// Transformer_spatial_10995116277909
// MI455X (gfx1250) — matrix operand order repaired and checked by reading, not compiled, not run on hardware
//
#include <hip/hip_runtime.h>
#include <hip/hip_bf16.h>

// ---------------------------------------------------------------------------
// Problem constants (from reference): B=8,S=32,F=64,C=512,D=2048,H=16,
// DK=DV=128, DINNER=8192.  Tokens per batch = C = 512, token dim = D = 2048.
// ---------------------------------------------------------------------------

typedef __attribute__((ext_vector_type(16))) _Float16 v16h;
typedef __attribute__((ext_vector_type(8)))  _Float16 v8h;
typedef __attribute__((ext_vector_type(8)))  float    v8f;

union AFrag { v16h v; v8h h[2]; };

#define WMMA_F16(a, b, c) \
  __builtin_amdgcn_wmma_f32_16x16x32_f16(false, (a), false, (b), (short)0, (c), false, false)

// ---------------------------------------------------------------------------
// Direct-from-global WMMA GEMM (used for the small per-(b,h) attention GEMMs).
// One wave per block computes a 32x32 tile as 2x2 WMMA 16x16x32 tiles.
// A row-major f16 [MxK], Bt = B^T row-major f16 [NxK].
// Fragment layouts per CDNA5 ISA tables:
//   A (16x32 f16): lane L: m = L&15, half = L>>4; elems 0..7  = K (8*half..)
//                                                elems 8..15 = K (16+8*half..)
//   B (32x16 f16) from Bt: lane L: n = L&15; elems 0..7 = K (8*half..), 8..15 = K (16+8*half..)
//   C/D (16x16 f32): lane L: n = L&15; VGPR r holds M = r + 8*(L>>4)
// ---------------------------------------------------------------------------
__global__ __launch_bounds__(32) void gemm_wmma_f16(
    const _Float16* __restrict__ A,  int lda, long a_bso, long a_bsi,
    const _Float16* __restrict__ Bt, int ldb, long b_bso, long b_bsi,
    float*          __restrict__ out32, long c32_bso, long c32_bsi,
    _Float16*       __restrict__ out16, long c16_bso, long c16_bsi,
    int ldc, int K,
    const float* __restrict__ bias,
    const float* __restrict__ resid, int ldr, long r_bso, long r_bsi,
    int inner_n, int relu)
{
  const int lane = threadIdx.x;                // 0..31, full wave, EXEC all 1s
  const int n0 = blockIdx.x * 32;
  const int m0 = blockIdx.y * 32;
  const int bo = blockIdx.z / inner_n;
  const int bi = blockIdx.z - bo * inner_n;
  A  += bo * a_bso + bi * a_bsi;
  Bt += bo * b_bso + bi * b_bsi;

  const int l16  = lane & 15;
  const int half = lane >> 4;

  const _Float16* pa0 = A + (long)(m0 + l16) * lda + 8 * half;
  const _Float16* pa1 = pa0 + (long)16 * lda;
  const _Float16* pb0 = Bt + (long)(n0 + l16) * ldb + 8 * half;
  const _Float16* pb1 = pb0 + (long)16 * ldb;

  v8f acc[2][2] = {};
  for (int k = 0; k < K; k += 32) {
    AFrag a0, a1;
    a0.h[0] = *(const v8h*)(pa0 + k);
    a0.h[1] = *(const v8h*)(pa0 + k + 16);
    a1.h[0] = *(const v8h*)(pa1 + k);
    a1.h[1] = *(const v8h*)(pa1 + k + 16);
    AFrag b0f, b1f;
    b0f.h[0] = *(const v8h*)(pb0 + k);
    b0f.h[1] = *(const v8h*)(pb0 + k + 16);
    b1f.h[0] = *(const v8h*)(pb1 + k);
    b1f.h[1] = *(const v8h*)(pb1 + k + 16);
    v16h b0 = b0f.v;
    v16h b1 = b1f.v;
    if (k + 128 < K) {                     // prefetch streaming operand 4 tiles ahead
      __builtin_prefetch(pb0 + k + 128, 0, 1);
      __builtin_prefetch(pb1 + k + 128, 0, 1);
    }
    acc[0][0] = WMMA_F16(a0.v, b0, acc[0][0]);
    acc[0][1] = WMMA_F16(a0.v, b1, acc[0][1]);
    acc[1][0] = WMMA_F16(a1.v, b0, acc[1][0]);
    acc[1][1] = WMMA_F16(a1.v, b1, acc[1][1]);
  }

  float*       o32 = out32 ? out32 + bo * c32_bso + bi * c32_bsi : nullptr;
  _Float16*    o16 = out16 ? out16 + bo * c16_bso + bi * c16_bsi : nullptr;
  const float* rs  = resid ? resid + bo * r_bso  + bi * r_bsi  : nullptr;

#pragma unroll
  for (int tm = 0; tm < 2; ++tm) {
#pragma unroll
    for (int tn = 0; tn < 2; ++tn) {
      const int   nn = n0 + tn * 16 + l16;
      const float bv = bias ? bias[nn] : 0.0f;
#pragma unroll
      for (int r = 0; r < 8; ++r) {
        const int mm = m0 + tm * 16 + 8 * half + r;
        float v = acc[tm][tn][r] + bv;
        if (rs)   v += rs[(long)mm * ldr + nn];
        if (relu) v = fmaxf(v, 0.0f);
        const long ci = (long)mm * ldc + nn;
        if (o32) o32[ci] = v;
        if (o16) o16[ci] = (_Float16)v;
      }
    }
  }
}

// ---------------------------------------------------------------------------
// LDS-staged WMMA GEMM for the big flat GEMMs (M,N multiples of 128, K of 32).
// 4 waves / block.  Block tile 128x128, each wave a 64x64 register tile
// (4x4 WMMA 16x16x32).  K is streamed in 32-wide chunks through a
// double-buffered LDS tile filled with CDNA5 async global->LDS copies
// (GLOBAL_LOAD_ASYNC_TO_LDS_B128, ASYNCcnt), giving ~64 FLOP per global byte.
// LDS rows padded to 40 halves (80 B) to spread bank accesses.
// Static LDS: 2 bufs * 2 mats * 128*40*2B = 40 KB.
// ---------------------------------------------------------------------------
__global__ __launch_bounds__(128, 1) void gemm_wmma_f16_lds(
    const _Float16* __restrict__ A,  int lda,
    const _Float16* __restrict__ Bt, int ldb,
    float*          __restrict__ out32,
    _Float16*       __restrict__ out16,
    int ldc, int K,
    const float* __restrict__ bias,
    const float* __restrict__ resid, int ldr, int relu)
{
  __shared__ _Float16 sA[2][128 * 40];
  __shared__ _Float16 sB[2][128 * 40];

  const int tid  = threadIdx.x;        // 0..127
  const int lane = tid & 31;
  const int wave = tid >> 5;           // 0..3 -> 2x2 over the 128x128 tile
  const int wm   = (wave >> 1) * 64;
  const int wn   = (wave & 1) * 64;
  const int l16  = lane & 15;
  const int half = lane >> 4;

  const int n0 = blockIdx.x * 128;
  const int m0 = blockIdx.y * 128;

  // LDS byte addresses for the async-copy destination VGPRs (low 32 bits of
  // the generic pointer are the LDS byte offset).
  const unsigned ldsA = (unsigned)(unsigned long long)(uintptr_t)&sA[0][0];
  const unsigned ldsB = (unsigned)(unsigned long long)(uintptr_t)&sB[0][0];
  const unsigned bufB = (unsigned)(128 * 40 * sizeof(_Float16));   // 10240 B

  // Cooperative copy mapping: 128x32 f16 tile = 512 x 16B units;
  // 128 threads x 4 units each.  unit u = t*128 + tid -> row u>>2, col (u&3)*8.
  const int crow = tid >> 2;           // 0..31 (+32*t)
  const int ccol = (tid & 3) * 8;      // element offset 0,8,16,24

  auto issue_copy = [&](int c, int buf) {
    const unsigned kb = (unsigned)c * 64u;            // chunk byte offset in K
    const unsigned la = ldsA + (unsigned)buf * bufB;
    const unsigned lb = ldsB + (unsigned)buf * bufB;
#pragma unroll
    for (int t = 0; t < 4; ++t) {
      const int r = crow + 32 * t;
      const unsigned lo = (unsigned)(r * 40 + ccol) * 2u;
      const unsigned ga = ((unsigned)(m0 + r) * (unsigned)lda + (unsigned)ccol) * 2u + kb;
      const unsigned gb = ((unsigned)(n0 + r) * (unsigned)ldb + (unsigned)ccol) * 2u + kb;
      asm volatile("global_load_async_to_lds_b128 %0, %1, %2"
                   :: "v"(la + lo), "v"(ga), "s"(A) : "memory");
      asm volatile("global_load_async_to_lds_b128 %0, %1, %2"
                   :: "v"(lb + lo), "v"(gb), "s"(Bt) : "memory");
    }
  };

  v8f acc[4][4] = {};
  const int nch = K >> 5;

  issue_copy(0, 0);
  for (int c = 0; c < nch; ++c) {
    const int buf = c & 1;
    asm volatile("s_wait_asynccnt 0x0" ::: "memory");  // my copies landed
    __syncthreads();                                   // everyone's copies landed
    if (c + 1 < nch) issue_copy(c + 1, buf ^ 1);       // prefetch next chunk

    AFrag af[4], bf[4];
#pragma unroll
    for (int i = 0; i < 4; ++i) {
      const int ar = wm + i * 16 + l16;
      af[i].h[0] = *(const v8h*)&sA[buf][ar * 40 + 8 * half];
      af[i].h[1] = *(const v8h*)&sA[buf][ar * 40 + 8 * half + 16];
      const int br = wn + i * 16 + l16;
      bf[i].h[0] = *(const v8h*)&sB[buf][br * 40 + 8 * half];
      bf[i].h[1] = *(const v8h*)&sB[buf][br * 40 + 8 * half + 16];
    }
#pragma unroll
    for (int i = 0; i < 4; ++i)
#pragma unroll
      for (int j = 0; j < 4; ++j)
        acc[i][j] = WMMA_F16(af[i].v, bf[j].v, acc[i][j]);
  }

  // ---- epilogue: bias / f32 residual / relu, dual f32 + f16 stores ----
#pragma unroll
  for (int ti = 0; ti < 4; ++ti) {
#pragma unroll
    for (int tj = 0; tj < 4; ++tj) {
      const int   nn = n0 + wn + tj * 16 + l16;
      const float bv = bias ? bias[nn] : 0.0f;
#pragma unroll
      for (int r = 0; r < 8; ++r) {
        const int mm = m0 + wm + ti * 16 + 8 * half + r;
        float v = acc[ti][tj][r] + bv;
        if (resid) v += resid[(long)mm * ldr + nn];
        if (relu)  v = fmaxf(v, 0.0f);
        const long ci = (long)mm * ldc + nn;
        if (out32) out32[ci] = v;
        if (out16) out16[ci] = (_Float16)v;
      }
    }
  }
}

// ---------------------------------------------------------------------------
// LDS-tiled transpose + convert:  in [rows x cols] -> out [cols x rows],
// out = (Tout)(scale * in).  Optional secondary f32 copy (same transposed
// layout) used to keep the f32 residual path.
// ---------------------------------------------------------------------------
template <typename Tin, typename Tout>
__global__ __launch_bounds__(256) void transpose_conv(
    const Tin* __restrict__ in, Tout* __restrict__ out, float* __restrict__ out32,
    int rows, int cols, long in_bs, long out_bs, float scale)
{
  __shared__ Tin tile[32][33];
  const int tx = threadIdx.x, ty = threadIdx.y;       // block (32,8)
  const long ib = (long)blockIdx.z * in_bs;
  const long ob = (long)blockIdx.z * out_bs;
  const int c0 = blockIdx.x * 32, r0 = blockIdx.y * 32;
#pragma unroll
  for (int t = 0; t < 4; ++t)
    tile[ty + t * 8][tx] = in[ib + (long)(r0 + ty + t * 8) * cols + (c0 + tx)];
  __syncthreads();
#pragma unroll
  for (int t = 0; t < 4; ++t) {
    float v = (float)tile[tx][ty + t * 8] * scale;
    long oi = ob + (long)(c0 + ty + t * 8) * rows + (r0 + tx);
    out[oi] = (Tout)v;
    if (out32) out32[oi] = v;
  }
}

// ---------------------------------------------------------------------------
// Row LayerNorm, in place on f32 rows of length Dlen, optional f16 copy out.
// ---------------------------------------------------------------------------
__global__ __launch_bounds__(256) void layernorm_rows(
    float* __restrict__ x, _Float16* __restrict__ out16,
    const float* __restrict__ g, const float* __restrict__ b, int Dlen)
{
  __shared__ float sh1[256], sh2[256];
  const long row = blockIdx.x;
  float* xr = x + row * (long)Dlen;
  const int tid = threadIdx.x;
  float s = 0.0f, s2 = 0.0f;
  for (int d = tid; d < Dlen; d += 256) { float v = xr[d]; s += v; s2 += v * v; }
  sh1[tid] = s; sh2[tid] = s2;
  __syncthreads();
  for (int o = 128; o > 0; o >>= 1) {
    if (tid < o) { sh1[tid] += sh1[tid + o]; sh2[tid] += sh2[tid + o]; }
    __syncthreads();
  }
  const float mean = sh1[0] / (float)Dlen;
  const float var  = sh2[0] / (float)Dlen - mean * mean;
  const float inv  = rsqrtf(var + 1e-6f);
  for (int d = tid; d < Dlen; d += 256) {
    float v = (xr[d] - mean) * inv * g[d] + b[d];
    xr[d] = v;
    if (out16) out16[row * (long)Dlen + d] = (_Float16)v;
  }
}

// ---------------------------------------------------------------------------
// Softmax over rows of 512, in place on f32 (d_out attn region) + f16 copy.
// ---------------------------------------------------------------------------
__global__ __launch_bounds__(256) void softmax512(
    float* __restrict__ attn, _Float16* __restrict__ attn16)
{
  __shared__ float smax[256], ssum[256];
  const long row = blockIdx.x;
  float* ar = attn + row * 512;
  const int t = threadIdx.x;
  const float v0 = ar[t], v1 = ar[t + 256];
  smax[t] = fmaxf(v0, v1);
  __syncthreads();
  for (int o = 128; o > 0; o >>= 1) {
    if (t < o) smax[t] = fmaxf(smax[t], smax[t + o]);
    __syncthreads();
  }
  const float mx = smax[0];
  const float e0 = __expf(v0 - mx), e1 = __expf(v1 - mx);
  ssum[t] = e0 + e1;
  __syncthreads();
  for (int o = 128; o > 0; o >>= 1) {
    if (t < o) ssum[t] += ssum[t + o];
    __syncthreads();
  }
  const float inv = 1.0f / ssum[0];
  const float r0 = e0 * inv, r1 = e1 * inv;
  ar[t] = r0; ar[t + 256] = r1;
  _Float16* a16 = attn16 + row * 512;
  a16[t] = (_Float16)r0; a16[t + 256] = (_Float16)r1;
}

// ---------------------------------------------------------------------------
// Host orchestration
// ---------------------------------------------------------------------------
extern "C" void kernel_launch(void* const* d_in, const int* in_sizes, int n_in,
                              void* d_out, int out_size, void* d_ws, size_t ws_size,
                              hipStream_t stream)
{
  (void)in_sizes; (void)n_in; (void)out_size; (void)ws_size;

  const float* x     = (const float*)d_in[0];   // [8,32,64,512] = [B, D, C] after reshape
  const float* v     = (const float*)d_in[1];
  const float* wq    = (const float*)d_in[2];   // [2048, 2048]
  const float* wk    = (const float*)d_in[3];
  const float* wv    = (const float*)d_in[4];
  const float* wfc   = (const float*)d_in[5];
  const float* ln1_g = (const float*)d_in[6];
  const float* ln1_b = (const float*)d_in[7];
  const float* w1    = (const float*)d_in[8];   // [2048, 8192]
  const float* b1    = (const float*)d_in[9];
  const float* w2    = (const float*)d_in[10];  // [8192, 2048]
  const float* b2    = (const float*)d_in[11];
  const float* ln2_g = (const float*)d_in[12];
  const float* ln2_b = (const float*)d_in[13];

  float* out_main = (float*)d_out;                       // [B,S,F,C] = 8,388,608 f32
  float* out_attn = out_main + (size_t)8388608;          // [B,H,512,512] f32

  // ---- workspace carve-out (256B aligned), ~448 MB total ----
  char* wsp = (char*)d_ws;
  auto take = [&](size_t bytes) -> void* {
    void* p = (void*)wsp;
    wsp += (bytes + 255) & ~(size_t)255;
    return p;
  };
  const size_t BC_D = (size_t)4096 * 2048;   // [B*C, D] elements
  _Float16* xt16   = (_Float16*)take(BC_D * 2);
  _Float16* vt16   = (_Float16*)take(BC_D * 2);
  float*    xt32   = (float*)   take(BC_D * 4);          // f32 residual copy
  _Float16* wqT16  = (_Float16*)take((size_t)2048 * 2048 * 2);
  _Float16* wkT16  = (_Float16*)take((size_t)2048 * 2048 * 2);
  _Float16* wvT16  = (_Float16*)take((size_t)2048 * 2048 * 2);
  _Float16* wfcT16 = (_Float16*)take((size_t)2048 * 2048 * 2);
  _Float16* w1T16  = (_Float16*)take((size_t)8192 * 2048 * 2);
  _Float16* w2T16  = (_Float16*)take((size_t)2048 * 8192 * 2);
  _Float16* q16    = (_Float16*)take(BC_D * 2);          // [B*C, H*DK]
  _Float16* k16    = (_Float16*)take(BC_D * 2);
  _Float16* val16  = (_Float16*)take(BC_D * 2);          // [B*C, H*DV]
  _Float16* valT16 = (_Float16*)take(BC_D * 2);          // [B, H*DV, C]
  _Float16* attn16 = (_Float16*)take((size_t)8 * 16 * 512 * 512 * 2);
  _Float16* o16    = (_Float16*)take(BC_D * 2);
  float*    o32    = (float*)   take(BC_D * 4);          // o projection -> LN1 (in place)
  _Float16* oln16  = (_Float16*)take(BC_D * 2);
  _Float16* h16    = (_Float16*)take((size_t)4096 * 8192 * 2);
  float*    y32    = (float*)   take(BC_D * 4);          // FFN out -> LN2 (in place)

  const dim3 tb(32, 8);

  // ---- input transposes:  [B, D=2048, C=512] f32 -> [B, C, D] f16 (+f32 for x) ----
  transpose_conv<float, _Float16><<<dim3(16, 64, 8), tb, 0, stream>>>(
      x, xt16, xt32, 2048, 512, 1048576, 1048576, 1.0f);
  transpose_conv<float, _Float16><<<dim3(16, 64, 8), tb, 0, stream>>>(
      v, vt16, nullptr, 2048, 512, 1048576, 1048576, 1.0f);

  // ---- weight convert+transpose to [N, K] f16; fold 1/sqrt(DK) into wq ----
  transpose_conv<float, _Float16><<<dim3(64, 64, 1), tb, 0, stream>>>(
      wq, wqT16, nullptr, 2048, 2048, 0, 0, 0.0883883476483184405f);
  transpose_conv<float, _Float16><<<dim3(64, 64, 1), tb, 0, stream>>>(
      wk, wkT16, nullptr, 2048, 2048, 0, 0, 1.0f);
  transpose_conv<float, _Float16><<<dim3(64, 64, 1), tb, 0, stream>>>(
      wv, wvT16, nullptr, 2048, 2048, 0, 0, 1.0f);
  transpose_conv<float, _Float16><<<dim3(64, 64, 1), tb, 0, stream>>>(
      wfc, wfcT16, nullptr, 2048, 2048, 0, 0, 1.0f);
  transpose_conv<float, _Float16><<<dim3(256, 64, 1), tb, 0, stream>>>(
      w1, w1T16, nullptr, 2048, 8192, 0, 0, 1.0f);
  transpose_conv<float, _Float16><<<dim3(64, 256, 1), tb, 0, stream>>>(
      w2, w2T16, nullptr, 8192, 2048, 0, 0, 1.0f);

  // ---- QKV projections via LDS/async GEMM: [4096,2048] x [2048,2048] ----
  gemm_wmma_f16_lds<<<dim3(16, 32, 1), dim3(128), 0, stream>>>(
      xt16, 2048, wqT16, 2048, nullptr, q16, 2048, 2048, nullptr, nullptr, 0, 0);
  gemm_wmma_f16_lds<<<dim3(16, 32, 1), dim3(128), 0, stream>>>(
      xt16, 2048, wkT16, 2048, nullptr, k16, 2048, 2048, nullptr, nullptr, 0, 0);
  gemm_wmma_f16_lds<<<dim3(16, 32, 1), dim3(128), 0, stream>>>(
      vt16, 2048, wvT16, 2048, nullptr, val16, 2048, 2048, nullptr, nullptr, 0, 0);

  // ---- v^T for attn@v:  [B, 512, 2048] f16 -> [B, 2048, 512] f16 ----
  transpose_conv<_Float16, _Float16><<<dim3(64, 16, 8), tb, 0, stream>>>(
      val16, valT16, nullptr, 512, 2048, 1048576, 1048576, 1.0f);

  // ---- scores = q k^T (scale folded) -> raw f32 straight into d_out attn ----
  // per (b,h): A base b*1048576 + h*128 (lda 2048), same for k; out base
  // b*4194304 + h*262144 (ldc 512).  M=N=512, K=128, 128 batches (inner=16).
  gemm_wmma_f16<<<dim3(16, 16, 128), dim3(32), 0, stream>>>(
      q16, 2048, 1048576, 128, k16, 2048, 1048576, 128,
      out_attn, 4194304, 262144, nullptr, 0, 0, 512, 128,
      nullptr, nullptr, 0, 0, 0, 16, 0);

  // ---- softmax in place on d_out attn + f16 copy for next GEMM ----
  softmax512<<<dim3(65536), dim3(256), 0, stream>>>(out_attn, attn16);

  // ---- o = attn @ v:  per (b,h) [512,512] x [512,128] -> o16 [B,C,H*DV] ----
  gemm_wmma_f16<<<dim3(4, 16, 128), dim3(32), 0, stream>>>(
      attn16, 512, 4194304, 262144, valT16, 512, 1048576, 65536,
      nullptr, 0, 0, o16, 1048576, 128, 2048, 512,
      nullptr, nullptr, 0, 0, 0, 16, 0);

  // ---- o @ wfc + residual(xt32) -> f32; then LN1 in place (+f16 copy) ----
  gemm_wmma_f16_lds<<<dim3(16, 32, 1), dim3(128), 0, stream>>>(
      o16, 2048, wfcT16, 2048, o32, nullptr, 2048, 2048, nullptr, xt32, 2048, 0);
  layernorm_rows<<<dim3(4096), dim3(256), 0, stream>>>(o32, oln16, ln1_g, ln1_b, 2048);

  // ---- FFN up: relu(o @ w1 + b1) -> h16 [4096, 8192] ----
  gemm_wmma_f16_lds<<<dim3(64, 32, 1), dim3(128), 0, stream>>>(
      oln16, 2048, w1T16, 2048, nullptr, h16, 8192, 2048, b1, nullptr, 0, 1);

  // ---- FFN down: h @ w2 + b2 + o (f32 residual) -> y32; LN2 in place ----
  gemm_wmma_f16_lds<<<dim3(16, 32, 1), dim3(128), 0, stream>>>(
      h16, 8192, w2T16, 8192, y32, nullptr, 2048, 8192, b2, o32, 2048, 0);
  layernorm_rows<<<dim3(4096), dim3(256), 0, stream>>>(y32, nullptr, ln2_g, ln2_b, 2048);

  // ---- final transpose: [B, C=512, D=2048] f32 -> d_out [B, D, C] = [B,S,F,C] ----
  transpose_conv<float, float><<<dim3(64, 16, 8), tb, 0, stream>>>(
      y32, out_main, nullptr, 512, 2048, 1048576, 1048576, 1.0f);
}
